// AgentAwareAttention_5076651344063
// MI455X (gfx1250) — matrix operand order repaired and checked by reading, not compiled, not run on hardware
//
#include <hip/hip_runtime.h>
#include <hip/hip_bf16.h>

// Problem constants (from reference)
#define LSEQ 2048
#define NB   2
#define EMB  256
#define NH   8
#define HD   32
#define BH   16      // NB*NH head-batches
#define MROWS 4096   // LSEQ*NB
#define SCALE_Q 0.17677669529663689f  // 32^-0.5

typedef __attribute__((ext_vector_type(16))) __bf16 bf16x16;
typedef __attribute__((ext_vector_type(8)))  __bf16 bf16x8;
typedef __attribute__((ext_vector_type(8)))  float  f32x8;
typedef __attribute__((ext_vector_type(4)))  float  f32x4;
typedef __attribute__((ext_vector_type(4)))  unsigned int u32x4;
typedef __attribute__((ext_vector_type(8)))  int    i32x8;
typedef __attribute__((ext_vector_type(4)))  int    i32x4;

#if __has_builtin(__builtin_amdgcn_tensor_load_to_lds)
#define HAVE_TDM 1
#else
#define HAVE_TDM 0
#endif

#if __has_builtin(__builtin_amdgcn_s_wait_tensorcnt)
#define WAIT_TENSORCNT(n) __builtin_amdgcn_s_wait_tensorcnt(n)
#else
#define WAIT_TENSORCNT(n) asm volatile("s_wait_tensorcnt %0" :: "i"(n) : "memory")
#endif

__device__ __forceinline__ bf16x16 cat8(bf16x8 lo, bf16x8 hi) {
    bf16x16 r;
#pragma unroll
    for (int i = 0; i < 8; ++i) { r[i] = lo[i]; r[i + 8] = hi[i]; }
    return r;
}

__device__ __forceinline__ f32x8 wmma_bf16(bf16x16 a, bf16x16 b, f32x8 c) {
    // D = A(16x32) * B(32x16) + C ; f32 accumulate
    return __builtin_amdgcn_wmma_f32_16x16x32_bf16(
        /*neg_a=*/false, a, /*neg_b=*/false, b,
        /*c_mod=*/(short)0, c, /*reuse_a=*/false, /*reuse_b=*/false);
}

// Load 8 consecutive f32 and convert to bf16x8
__device__ __forceinline__ bf16x8 cvt8(const float* __restrict__ p) {
    f32x4 u0 = *(const f32x4*)(p);
    f32x4 u1 = *(const f32x4*)(p + 4);
    bf16x8 r;
#pragma unroll
    for (int i = 0; i < 4; ++i) { r[i] = (__bf16)u0[i]; r[i + 4] = (__bf16)u1[i]; }
    return r;
}

#if HAVE_TDM
// Issue one TDM 2-D tile load (bf16 elements) into LDS.
// Descriptor bitfields per CDNA5 ISA:
//   group0: count=1 | lds_addr | global_addr[56:0] | type=2
//   group1: data_size=1 (2B), tensor_dim0/1, tile_dim0/1, tensor_dim0_stride
__device__ __forceinline__ void tdm_load_2d_bf16(
    const __bf16* gptr, unsigned lds_off,
    unsigned tdim0, unsigned tdim1,
    unsigned tile0, unsigned tile1, unsigned stride0)
{
    unsigned long long ga = (unsigned long long)gptr;
    u32x4 g0;
    g0[0] = 1u;                                   // count=1, user mode
    g0[1] = lds_off;                              // lds_addr (bytes)
    g0[2] = (unsigned)ga;                         // global_addr[31:0]
    g0[3] = (unsigned)((ga >> 32) & 0x01FFFFFFu)  // global_addr[56:32]
            | (2u << 30);                         // type=2 ("image")
    i32x8 g1;
    g1[0] = (int)(1u << 16);                      // workgroup_mask=0, data_size=1 (2B)
    g1[1] = (int)((tdim0 & 0xFFFFu) << 16);       // tensor_dim0[15:0]
    g1[2] = (int)(((tdim0 >> 16) & 0xFFFFu) | ((tdim1 & 0xFFFFu) << 16));
    g1[3] = (int)(((tdim1 >> 16) & 0xFFFFu) | ((tile0 & 0xFFFFu) << 16));
    g1[4] = (int)(tile1 & 0xFFFFu);               // tile_dim1, tile_dim2=0
    g1[5] = (int)stride0;                         // tensor_dim0_stride[31:0]
    g1[6] = 0;                                    // stride0[47:32], stride1[15:0]
    g1[7] = 0;                                    // stride1[47:16]
    i32x4 z4 = {0, 0, 0, 0};
#if __clang_major__ >= 23
    i32x8 z8 = {0, 0, 0, 0, 0, 0, 0, 0};
    __builtin_amdgcn_tensor_load_to_lds(g0, g1, z4, z4, z8, 0);
#else
    __builtin_amdgcn_tensor_load_to_lds(g0, g1, z4, z4, 0);
#endif
}

// Low 32 bits of a flat LDS pointer are the LDS byte offset (ISA §10.2 aperture).
__device__ __forceinline__ unsigned lds_off_of(const void* p) {
    return (unsigned)(unsigned long long)p;
}
#endif

// ---------------------------------------------------------------------------
// Kernel 1: fused input projections.
//   [qkv | qk_self] = X (4096x256) @ [Wqkv;Wself]^T (256x1280) + bias
// One 16x16 output tile per wave; K-loop over 256 in chunks of 32.
// Scatter-stores into head-major bf16 buffers; V stored transposed (D x L).
// ---------------------------------------------------------------------------
__global__ __launch_bounds__(256) void proj_kernel(
    const float* __restrict__ X,
    const float* __restrict__ Wqkv,  const float* __restrict__ bqkv,
    const float* __restrict__ Wself, const float* __restrict__ bself,
    __bf16* __restrict__ Qh, __bf16* __restrict__ Kh, __bf16* __restrict__ Vt,
    __bf16* __restrict__ Qs, __bf16* __restrict__ Ks)
{
    const int wave = threadIdx.x >> 5;
    const int lid  = threadIdx.x & 31;
    const int n = lid & 15;       // A-row within tile / B-col within tile
    const int h = lid >> 4;       // lane half

    const int JT = 80;            // 1280/16 column tiles
    int tile  = blockIdx.x * 8 + wave;   // 2560 blocks * 8 waves = 20480 tiles
    int mt    = tile / JT, jt = tile % JT;
    int mbase = mt * 16, jbase = jt * 16;
    int jg    = jbase + n;        // global output column for this lane

    const float* Wrow; float bias;
    if (jg < 768) { Wrow = Wqkv  + jg * EMB;         bias = bqkv[jg]; }
    else          { Wrow = Wself + (jg - 768) * EMB; bias = bself[jg - 768]; }

    const float* Arow = X + (mbase + n) * EMB;

    f32x8 c = {0.f, 0.f, 0.f, 0.f, 0.f, 0.f, 0.f, 0.f};
#pragma unroll
    for (int kc = 0; kc < 8; ++kc) {
        int kb = kc * 32;
        bf16x16 a = cat8(cvt8(Arow + kb + 8 * h),
                         cvt8(Arow + kb + 16 + 8 * h));
        bf16x16 b = cat8(cvt8(Wrow + kb + 8 * h),
                         cvt8(Wrow + kb + 16 + 8 * h));
        c = wmma_bf16(a, b, c);
    }

    int seg = jbase >> 8;               // 0=q 1=k 2=v 3=q_self 4=k_self (uniform/tile)
    int jj  = jg & 255;
    int hh  = jj >> 5, d = jj & 31;     // head, dim within head
    float scale = (seg == 0 || seg == 3) ? SCALE_Q : 1.0f;

#pragma unroll
    for (int r = 0; r < 8; ++r) {
        int m  = mbase + r + 8 * h;     // C/D layout: row = r + 8h
        int l  = m >> 1, nb = m & 1;    // X row = l*NB + nb
        int bi = nb * NH + hh;          // head-batch index
        __bf16 bv = (__bf16)((c[r] + bias) * scale);
        switch (seg) {
            case 0:  Qh[(bi * LSEQ + l) * HD + d] = bv; break;
            case 1:  Kh[(bi * LSEQ + l) * HD + d] = bv; break;
            case 2:  Vt[(bi * HD + d) * LSEQ + l] = bv; break;  // transposed
            case 3:  Qs[(bi * LSEQ + l) * HD + d] = bv; break;
            default: Ks[(bi * LSEQ + l) * HD + d] = bv; break;
        }
    }
}

// ---------------------------------------------------------------------------
// Kernel 2: fused agent-aware flash attention.
// Grid = BH * (L/64) blocks, 4 waves/block; each wave owns 16 query rows.
// K / K_self / V^T tiles (32 keys) are staged once per workgroup into LDS by
// the Tensor Data Mover (double-buffered, TENSORcnt), then all 4 waves run
// 4 QK WMMAs + mask select + online softmax + P relayout + 2 AV WMMAs.
// ---------------------------------------------------------------------------
__global__ __launch_bounds__(128) void attn_kernel(
    const __bf16* __restrict__ Qh, const __bf16* __restrict__ Kh,
    const __bf16* __restrict__ Vt, const __bf16* __restrict__ Qs,
    const __bf16* __restrict__ Ks,
    const int* __restrict__ qid, const int* __restrict__ kid,
    float* __restrict__ Obuf)
{
#if HAVE_TDM
    __shared__ __align__(128) __bf16 KtileL[2][32 * 32];   // [key][d]
    __shared__ __align__(128) __bf16 KstileL[2][32 * 32];  // [key][d]
    __shared__ __align__(128) __bf16 VtileL[2][32 * 32];   // [d][key]
#endif
    __shared__ __align__(64) __bf16 Pbuf[4][16 * 32];

    const int wave = threadIdx.x >> 5;
    const int lid  = threadIdx.x & 31;
    const int n = lid & 15;
    const int h = lid >> 4;

    const int bi    = blockIdx.x >> 5;          // head-batch 0..15
    const int chunk = blockIdx.x & 31;          // 64-row chunk
    const int lbase = chunk * 64 + wave * 16;

    const __bf16* Qb  = Qh + (size_t)bi * LSEQ * HD;
    const __bf16* Kb  = Kh + (size_t)bi * LSEQ * HD;
    const __bf16* Qsb = Qs + (size_t)bi * LSEQ * HD;
    const __bf16* Ksb = Ks + (size_t)bi * LSEQ * HD;
    const __bf16* Vb  = Vt + (size_t)bi * HD * LSEQ;

    // A operands: Q and Q_self fragments for these 16 rows (row = n per A layout)
    bf16x16 qa  = cat8(*(const bf16x8*)(Qb  + (lbase + n) * HD + 16 * h),
                       *(const bf16x8*)(Qb  + (lbase + n) * HD + 16 * h + 8));
    bf16x16 qsa = cat8(*(const bf16x8*)(Qsb + (lbase + n) * HD + 16 * h),
                       *(const bf16x8*)(Qsb + (lbase + n) * HD + 16 * h + 8));

    int qidr[8];
#pragma unroll
    for (int r = 0; r < 8; ++r) qidr[r] = qid[lbase + r + 8 * h];

    float rowmax[8], rowsum[8];
#pragma unroll
    for (int r = 0; r < 8; ++r) { rowmax[r] = -1e30f; rowsum[r] = 0.f; }
    f32x8 O0 = {0.f, 0.f, 0.f, 0.f, 0.f, 0.f, 0.f, 0.f};
    f32x8 O1 = O0;

    __bf16* P = &Pbuf[wave][0];

#if HAVE_TDM
    // Prologue: stage tiles for st=0 into buffer 0 (wave 0 issues for the WG;
    // TDM ignores EXEC, one instruction per wave).
    if (wave == 0) {
        tdm_load_2d_bf16(Kb,       lds_off_of(&KtileL[0][0]),  HD, LSEQ, 32, 32, HD);
        tdm_load_2d_bf16(Ksb,      lds_off_of(&KstileL[0][0]), HD, LSEQ, 32, 32, HD);
        tdm_load_2d_bf16(Vb,       lds_off_of(&VtileL[0][0]),  LSEQ, HD, 32, 32, LSEQ);
    }
#endif

    for (int st = 0; st < 64; ++st) {
        int sb = st * 32;
#if HAVE_TDM
        int cur = st & 1;
        if (wave == 0) {
            if (st + 1 < 64) {
                int nsb = sb + 32, nxt = cur ^ 1;
                tdm_load_2d_bf16(Kb  + (size_t)nsb * HD, lds_off_of(&KtileL[nxt][0]),  HD, LSEQ, 32, 32, HD);
                tdm_load_2d_bf16(Ksb + (size_t)nsb * HD, lds_off_of(&KstileL[nxt][0]), HD, LSEQ, 32, 32, HD);
                tdm_load_2d_bf16(Vb  + nsb,              lds_off_of(&VtileL[nxt][0]),  LSEQ, HD, 32, 32, LSEQ);
                WAIT_TENSORCNT(3);      // current tile's 3 loads retired
            } else {
                WAIT_TENSORCNT(0);
            }
        }
        __syncthreads();                // tiles for st visible to all waves

        const __bf16* Kl  = &KtileL[cur][0];
        const __bf16* Ksl = &KstileL[cur][0];
        const __bf16* Vl  = &VtileL[cur][0];
        bf16x16 kb0  = *(const bf16x16*)(Kl  + n        * HD + 16 * h);
        bf16x16 kb1  = *(const bf16x16*)(Kl  + (16 + n) * HD + 16 * h);
        bf16x16 ksb0 = *(const bf16x16*)(Ksl + n        * HD + 16 * h);
        bf16x16 ksb1 = *(const bf16x16*)(Ksl + (16 + n) * HD + 16 * h);
        bf16x16 vb0  = *(const bf16x16*)(Vl  + n        * 32 + 16 * h);
        bf16x16 vb1  = *(const bf16x16*)(Vl  + (16 + n) * 32 + 16 * h);
#else
        bf16x16 kb0  = *(const bf16x16*)(Kb  + (sb + n)      * HD + 16 * h);
        bf16x16 kb1  = *(const bf16x16*)(Kb  + (sb + 16 + n) * HD + 16 * h);
        bf16x16 ksb0 = *(const bf16x16*)(Ksb + (sb + n)      * HD + 16 * h);
        bf16x16 ksb1 = *(const bf16x16*)(Ksb + (sb + 16 + n) * HD + 16 * h);
        bf16x16 vb0  = *(const bf16x16*)(Vb + n        * LSEQ + sb + 16 * h);
        bf16x16 vb1  = *(const bf16x16*)(Vb + (16 + n) * LSEQ + sb + 16 * h);
#endif

        f32x8 z = {0.f, 0.f, 0.f, 0.f, 0.f, 0.f, 0.f, 0.f};
        f32x8 ci0 = wmma_bf16(qa,  kb0,  z);
        f32x8 ci1 = wmma_bf16(qa,  kb1,  z);
        f32x8 cs0 = wmma_bf16(qsa, ksb0, z);
        f32x8 cs1 = wmma_bf16(qsa, ksb1, z);

        int kid0 = kid[sb + n];
        int kid1 = kid[sb + 16 + n];

#pragma unroll
        for (int r = 0; r < 8; ++r) {
            // agent-aware select: self score where identities match
            float s0 = (kid0 == qidr[r]) ? cs0[r] : ci0[r];
            float s1 = (kid1 == qidr[r]) ? cs1[r] : ci1[r];
            // online softmax: row spans the 16 lanes of this half
            float tm = fmaxf(s0, s1);
#pragma unroll
            for (int off = 1; off < 16; off <<= 1)
                tm = fmaxf(tm, __shfl_xor(tm, off, 32));
            float nm    = fmaxf(rowmax[r], tm);
            float alpha = __expf(rowmax[r] - nm);
            float p0    = __expf(s0 - nm);
            float p1    = __expf(s1 - nm);
            float ts    = p0 + p1;
#pragma unroll
            for (int off = 1; off < 16; off <<= 1)
                ts += __shfl_xor(ts, off, 32);
            rowsum[r] = rowsum[r] * alpha + ts;
            rowmax[r] = nm;
            O0[r] *= alpha;
            O1[r] *= alpha;
            // stash probabilities (C/D layout -> LDS row-major [m][k])
            P[(r + 8 * h) * 32 + n]      = (__bf16)p0;
            P[(r + 8 * h) * 32 + 16 + n] = (__bf16)p1;
        }

        // DS ops are in-order per wave; wait for stores before cross-lane reads
        asm volatile("s_wait_dscnt 0" ::: "memory");

        // Reload P in A layout (row = n, contiguous 16-elem run at 16h)
        bf16x16 pa = cat8(*(const bf16x8*)(P + n * 32 + 16 * h),
                          *(const bf16x8*)(P + n * 32 + 16 * h + 8));
        O0 = wmma_bf16(pa, vb0, O0);
        O1 = wmma_bf16(pa, vb1, O1);

#if HAVE_TDM
        __syncthreads();                // all reads of buf[cur] done before reuse
#endif
    }

    const int nb = bi >> 3, hh = bi & 7;
#pragma unroll
    for (int r = 0; r < 8; ++r) {
        float inv = 1.0f / rowsum[r];
        int l = lbase + r + 8 * h;
        float* orow = Obuf + ((size_t)l * NB + nb) * EMB + hh * HD;
        orow[n]      = O0[r] * inv;
        orow[16 + n] = O1[r] * inv;
    }
}

// ---------------------------------------------------------------------------
// Kernel 3: output projection  Y = Obuf(4096x256) @ Wo^T + bo
// ---------------------------------------------------------------------------
__global__ __launch_bounds__(256) void outproj_kernel(
    const float* __restrict__ Obuf, const float* __restrict__ Wo,
    const float* __restrict__ bo, float* __restrict__ Y)
{
    const int wave = threadIdx.x >> 5;
    const int lid  = threadIdx.x & 31;
    const int n = lid & 15;
    const int h = lid >> 4;

    int tile  = blockIdx.x * 8 + wave;   // 512 blocks * 8 = 4096 tiles
    int mt    = tile >> 4, jt = tile & 15;
    int mbase = mt * 16, jbase = jt * 16;

    const float* Arow = Obuf + (mbase + n) * EMB;
    const float* Wrow = Wo + (jbase + n) * EMB;
    float bias = bo[jbase + n];

    f32x8 c = {0.f, 0.f, 0.f, 0.f, 0.f, 0.f, 0.f, 0.f};
#pragma unroll
    for (int kc = 0; kc < 8; ++kc) {
        int kb = kc * 32;
        bf16x16 a = cat8(cvt8(Arow + kb + 8 * h),
                         cvt8(Arow + kb + 16 + 8 * h));
        bf16x16 b = cat8(cvt8(Wrow + kb + 8 * h),
                         cvt8(Wrow + kb + 16 + 8 * h));
        c = wmma_bf16(a, b, c);
    }

#pragma unroll
    for (int r = 0; r < 8; ++r)
        Y[(size_t)(mbase + r + 8 * h) * EMB + jbase + n] = c[r] + bias;
}

// ---------------------------------------------------------------------------
extern "C" void kernel_launch(void* const* d_in, const int* in_sizes, int n_in,
                              void* d_out, int out_size, void* d_ws, size_t ws_size,
                              hipStream_t stream)
{
    const float* X     = (const float*)d_in[0];
    const float* Wqkv  = (const float*)d_in[1];
    const float* bqkv  = (const float*)d_in[2];
    const float* Wself = (const float*)d_in[3];
    const float* bself = (const float*)d_in[4];
    const float* Wo    = (const float*)d_in[5];
    const float* bo    = (const float*)d_in[6];
    const int*   qid   = (const int*)d_in[7];
    const int*   kid   = (const int*)d_in[8];

    char* ws = (char*)d_ws;
    __bf16* Qh = (__bf16*)(ws + (0u  << 20));   // 2 MB each, bf16 head-major
    __bf16* Kh = (__bf16*)(ws + (2u  << 20));
    __bf16* Vt = (__bf16*)(ws + (4u  << 20));   // V transposed: [BH][HD][L]
    __bf16* Qs = (__bf16*)(ws + (6u  << 20));
    __bf16* Ks = (__bf16*)(ws + (8u  << 20));
    float*  Ob = (float*)(ws + (10u << 20));    // 4 MB f32, (L,N,E)

    proj_kernel<<<dim3(2560), dim3(256), 0, stream>>>(
        X, Wqkv, bqkv, Wself, bself, Qh, Kh, Vt, Qs, Ks);
    attn_kernel<<<dim3(512), dim3(128), 0, stream>>>(
        Qh, Kh, Vt, Qs, Ks, qid, kid, Ob);
    outproj_kernel<<<dim3(512), dim3(256), 0, stream>>>(
        Ob, Wo, bo, (float*)d_out);
}
